// GNN_73237782332046
// MI455X (gfx1250) — matrix operand order repaired and checked by reading, not compiled, not run on hardware
//
#include <hip/hip_runtime.h>

// Problem constants (fixed by the reference)
#define NSRV 10000
#define NDEV 50000
#define HDIM 512
#define NEDG 300000
#define NLAY 4

typedef __attribute__((ext_vector_type(4)))  __bf16 v4bf;
typedef __attribute__((ext_vector_type(8)))  __bf16 v8bf;
typedef __attribute__((ext_vector_type(16))) __bf16 v16bf;
typedef __attribute__((ext_vector_type(8)))  float  v8f;

// -------------------------------------------------------------------------
// zero helper (graph-capture-safe)
// -------------------------------------------------------------------------
__global__ void zero_f4(float4* __restrict__ p, long n4) {
  long i = (long)blockIdx.x * blockDim.x + threadIdx.x;
  if (i < n4) p[i] = make_float4(0.f, 0.f, 0.f, 0.f);
}

// -------------------------------------------------------------------------
// Weight prep: f32 [K][N] row-major -> bf16 [N][K] (transposed) so a WMMA
// B-fragment (column n, contiguous K) is two 16B loads per lane.
// grid: (H/32, H/32, nmat); block (32,8)
// -------------------------------------------------------------------------
__global__ void transpose_w(const float* __restrict__ W, __bf16* __restrict__ WT) {
  __shared__ float tile[32][33];
  const float* Wm = W + (size_t)blockIdx.z * HDIM * HDIM;
  __bf16* WTm = WT + (size_t)blockIdx.z * HDIM * HDIM;
  int n0 = blockIdx.x * 32, k0 = blockIdx.y * 32;
  for (int i = threadIdx.y; i < 32; i += 8)
    tile[i][threadIdx.x] = Wm[(size_t)(k0 + i) * HDIM + n0 + threadIdx.x];
  __syncthreads();
  for (int i = threadIdx.y; i < 32; i += 8)
    WTm[(size_t)(n0 + i) * HDIM + k0 + threadIdx.x] = (__bf16)tile[threadIdx.x][i];
}

// -------------------------------------------------------------------------
// Input projection: h = X @ W + b  (K = 64 or 32, tiny: plain VALU is fine)
// -------------------------------------------------------------------------
__global__ void input_proj(const float* __restrict__ X, const float* __restrict__ W,
                           const float* __restrict__ b, __bf16* __restrict__ out, int K) {
  __shared__ float xrow[64];
  int row = blockIdx.x, col = threadIdx.x;
  if (col < K) xrow[col] = X[(size_t)row * K + col];
  __syncthreads();
  float acc = b[col];
  for (int k = 0; k < K; ++k) acc = fmaf(xrow[k], W[(size_t)k * HDIM + col], acc);
  out[(size_t)row * HDIM + col] = (__bf16)acc;
}

// -------------------------------------------------------------------------
// Edge scatter: agg[dst] += h[src] (f32 atomics; agg buffers are L2-resident),
// cnt[dst] += 1 once per edge. block = 128 threads (4 feats each), grid = E.
// -------------------------------------------------------------------------
__global__ void scatter_accum(const __bf16* __restrict__ h, const int* __restrict__ esrc,
                              const int* __restrict__ edst, float* __restrict__ agg,
                              float* __restrict__ cnt) {
  int e = blockIdx.x;
  int s = esrc[e], d = edst[e];
  int f = threadIdx.x * 4;
  v4bf v = *(const v4bf*)(h + (size_t)s * HDIM + f);
  float* out = agg + (size_t)d * HDIM + f;
  atomicAdd(out + 0, (float)v[0]);
  atomicAdd(out + 1, (float)v[1]);
  atomicAdd(out + 2, (float)v[2]);
  atomicAdd(out + 3, (float)v[3]);
  if (threadIdx.x == 0) atomicAdd(cnt + d, 1.0f);
}

// agg[i][:] /= max(cnt[i],1); exact grid: N*H/4 threads
__global__ void normalize_agg(float* __restrict__ agg, const float* __restrict__ cnt) {
  size_t i = (size_t)blockIdx.x * blockDim.x + threadIdx.x;
  size_t row = i >> 7;                       // H/4 = 128 float4 per row
  float s = 1.0f / fmaxf(cnt[row], 1.0f);
  float4* p = (float4*)agg + i;
  float4 v = *p;
  v.x *= s; v.y *= s; v.z *= s; v.w *= s;
  *p = v;
}

// -------------------------------------------------------------------------
// Fused SAGE layer GEMM:  out = relu(Aagg @ Wl + bias + Ah @ Wr)
//   Aagg: [M,512] f32 (converted to bf16 while staging LDS A-chunk)
//   Ah:   [M,512] bf16
//   BTl/BTr: bf16 [512 n][512 k] (pre-transposed weights, L2-resident)
// block = 128 thr = 4 waves; block tile = 16 rows x 256 cols (wave: 16x64).
// A staged in 16x128 LDS chunks (8 barrier pairs total); per 32-wide K
// substep all 8 B-loads are issued before the 4 WMMAs to allow partial
// loadcnt waits / pipelining. WMMA fragment layouts per CDNA5 ISA 7.12.2.
// -------------------------------------------------------------------------
#define KCH   128          // K elements staged per barrier window
#define LDSA  136          // LDS row stride (bf16): 272B = 17*16B, 16B-aligned
__global__ __launch_bounds__(128) void sage_gemm(
    const float*  __restrict__ Aagg,
    const __bf16* __restrict__ Ah,
    const __bf16* __restrict__ BTl,
    const __bf16* __restrict__ BTr,
    const float*  __restrict__ bias,
    float*        __restrict__ out_f32,   // last layer only
    __bf16*       __restrict__ out_bf)    // hidden layers only
{
  __shared__ __bf16 As[16 * LDSA];
  const int tid  = threadIdx.x;
  const int wave = tid >> 5, lane = tid & 31;
  const int m    = lane & 15, half = lane >> 4;
  const int row0 = blockIdx.x * 16;
  const int ncol0 = blockIdx.y * 256 + wave * 64;
  const int sr = tid >> 3;            // staging row 0..15
  const int sk = (tid & 7) * 16;      // staging k-offset within chunk (32B)

  v8f c[4] = {};

  // 16 wmmas on the staged 16x128 chunk starting at global K = kc
  auto mma_chunk = [&](const __bf16* __restrict__ BT, int kc) {
    #pragma unroll
    for (int ks = 0; ks < 4; ++ks) {
      const __bf16* ab = &As[m * LDSA + ks * 32 + half * 8];
      v8bf alo = *(const v8bf*)ab;
      v8bf ahi = *(const v8bf*)(ab + 16);
      v16bf a = __builtin_shufflevector(alo, ahi,
                  0,1,2,3,4,5,6,7,8,9,10,11,12,13,14,15);
      v16bf bfrag[4];
      #pragma unroll
      for (int t = 0; t < 4; ++t) {   // issue all B loads up-front
        const __bf16* bp = BT + (size_t)(ncol0 + t * 16 + m) * HDIM
                              + kc + ks * 32 + half * 8;
        v8bf blo = *(const v8bf*)bp;
        v8bf bhi = *(const v8bf*)(bp + 16);
        bfrag[t] = __builtin_shufflevector(blo, bhi,
                     0,1,2,3,4,5,6,7,8,9,10,11,12,13,14,15);
      }
      #pragma unroll
      for (int t = 0; t < 4; ++t)
        c[t] = __builtin_amdgcn_wmma_f32_16x16x32_bf16(
                   false, a, false, bfrag[t], (short)0, c[t], false, false);
    }
  };

  // ---- source 1: agg (f32 -> bf16 while staging), weights Wl ----
  #pragma unroll 1
  for (int kc = 0; kc < HDIM; kc += KCH) {
    const float* ap = Aagg + (size_t)(row0 + sr) * HDIM + kc + sk;
    float4 v0 = ((const float4*)ap)[0];
    float4 v1 = ((const float4*)ap)[1];
    float4 v2 = ((const float4*)ap)[2];
    float4 v3 = ((const float4*)ap)[3];
    v8bf w0, w1;
    w0[0]=(__bf16)v0.x; w0[1]=(__bf16)v0.y; w0[2]=(__bf16)v0.z; w0[3]=(__bf16)v0.w;
    w0[4]=(__bf16)v1.x; w0[5]=(__bf16)v1.y; w0[6]=(__bf16)v1.z; w0[7]=(__bf16)v1.w;
    w1[0]=(__bf16)v2.x; w1[1]=(__bf16)v2.y; w1[2]=(__bf16)v2.z; w1[3]=(__bf16)v2.w;
    w1[4]=(__bf16)v3.x; w1[5]=(__bf16)v3.y; w1[6]=(__bf16)v3.z; w1[7]=(__bf16)v3.w;
    __bf16* dp = &As[sr * LDSA + sk];
    *(v8bf*)dp = w0;
    *(v8bf*)(dp + 8) = w1;
    __syncthreads();
    mma_chunk(BTl, kc);
    __syncthreads();
  }

  // ---- source 2: h (bf16 copy), weights Wr ----
  #pragma unroll 1
  for (int kc = 0; kc < HDIM; kc += KCH) {
    const __bf16* ap = Ah + (size_t)(row0 + sr) * HDIM + kc + sk;
    v8bf w0 = ((const v8bf*)ap)[0];
    v8bf w1 = ((const v8bf*)ap)[1];
    __bf16* dp = &As[sr * LDSA + sk];
    *(v8bf*)dp = w0;
    *(v8bf*)(dp + 8) = w1;
    __syncthreads();
    mma_chunk(BTr, kc);
    __syncthreads();
  }

  // ---- epilogue: bias + relu; D layout: M = half*8 + v, N = lane&15 ----
  #pragma unroll
  for (int t = 0; t < 4; ++t) {
    int col = ncol0 + t * 16 + m;
    float bv = bias[col];
    #pragma unroll
    for (int v = 0; v < 8; ++v) {
      float val = c[t][v] + bv;
      val = val > 0.f ? val : 0.f;
      size_t idx = (size_t)(row0 + half * 8 + v) * HDIM + col;
      if (out_f32) out_f32[idx] = val;
      if (out_bf)  out_bf[idx]  = (__bf16)val;
    }
  }
}

// -------------------------------------------------------------------------
extern "C" void kernel_launch(void* const* d_in, const int* in_sizes, int n_in,
                              void* d_out, int out_size, void* d_ws, size_t ws_size,
                              hipStream_t stream) {
  const float* x_srv = (const float*)d_in[0];
  const float* x_dev = (const float*)d_in[1];
  const float* W_srv = (const float*)d_in[2];
  const float* b_srv = (const float*)d_in[3];
  const float* W_dev = (const float*)d_in[4];
  const float* b_dev = (const float*)d_in[5];
  const float* Wl    = (const float*)d_in[6];
  const float* bl    = (const float*)d_in[7];
  const float* Wr    = (const float*)d_in[8];
  const int* e_sd_src = (const int*)d_in[9];
  const int* e_sd_dst = (const int*)d_in[10];
  const int* e_ds_src = (const int*)d_in[11];
  const int* e_ds_dst = (const int*)d_in[12];

  char* ws = (char*)d_ws;
  size_t off = 0;
  auto take = [&](size_t bytes) -> char* {
    char* p = ws + off;
    off += (bytes + 255) & ~(size_t)255;
    return p;
  };
  // activations as bf16 ping-pong; aggregates in f32 (L2-resident)
  __bf16* hs_bf[2] = { (__bf16*)take((size_t)NSRV * HDIM * 2),
                       (__bf16*)take((size_t)NSRV * HDIM * 2) };
  __bf16* hd_bf[2] = { (__bf16*)take((size_t)NDEV * HDIM * 2),
                       (__bf16*)take((size_t)NDEV * HDIM * 2) };
  float*  agg_s = (float*)take((size_t)NSRV * HDIM * 4);
  float*  agg_d = (float*)take((size_t)NDEV * HDIM * 4);
  float*  cnt_s = (float*)take((size_t)NSRV * 4);
  float*  cnt_d = (float*)take((size_t)NDEV * 4);
  __bf16* WlT = (__bf16*)take((size_t)NLAY * 2 * HDIM * HDIM * 2);
  __bf16* WrT = (__bf16*)take((size_t)NLAY * 2 * HDIM * HDIM * 2);

  // one-time (per launch) weight convert+transpose and input projection
  transpose_w<<<dim3(HDIM/32, HDIM/32, NLAY*2), dim3(32, 8), 0, stream>>>(Wl, WlT);
  transpose_w<<<dim3(HDIM/32, HDIM/32, NLAY*2), dim3(32, 8), 0, stream>>>(Wr, WrT);
  input_proj<<<NSRV, HDIM, 0, stream>>>(x_srv, W_srv, b_srv, hs_bf[0], 64);
  input_proj<<<NDEV, HDIM, 0, stream>>>(x_dev, W_dev, b_dev, hd_bf[0], 32);

  float* out_hs = (float*)d_out;                       // [NS,H]
  float* out_hd = (float*)d_out + (size_t)NSRV * HDIM; // [ND,H]

  int cur = 0;
  for (int l = 0; l < NLAY; ++l) {
    long n4;
    n4 = (long)NDEV * HDIM / 4;
    zero_f4<<<(n4 + 255) / 256, 256, 0, stream>>>((float4*)agg_d, n4);
    n4 = (long)NSRV * HDIM / 4;
    zero_f4<<<(n4 + 255) / 256, 256, 0, stream>>>((float4*)agg_s, n4);
    n4 = NDEV / 4;
    zero_f4<<<(n4 + 255) / 256, 256, 0, stream>>>((float4*)cnt_d, n4);
    n4 = NSRV / 4;
    zero_f4<<<(n4 + 255) / 256, 256, 0, stream>>>((float4*)cnt_s, n4);

    scatter_accum<<<NEDG, 128, 0, stream>>>(hs_bf[cur], e_sd_src, e_sd_dst, agg_d, cnt_d);
    scatter_accum<<<NEDG, 128, 0, stream>>>(hd_bf[cur], e_ds_src, e_ds_dst, agg_s, cnt_s);
    normalize_agg<<<NDEV / 2, 256, 0, stream>>>(agg_d, cnt_d);
    normalize_agg<<<NSRV / 2, 256, 0, stream>>>(agg_s, cnt_s);

    const bool last = (l == NLAY - 1);
    const int nxt = cur ^ 1;
    sage_gemm<<<dim3(NDEV / 16, 2), 128, 0, stream>>>(
        agg_d, hd_bf[cur],
        WlT + (size_t)(l * 2 + 0) * HDIM * HDIM,
        WrT + (size_t)(l * 2 + 0) * HDIM * HDIM,
        bl + (size_t)(l * 2 + 0) * HDIM,
        last ? out_hd : nullptr, last ? nullptr : hd_bf[nxt]);
    sage_gemm<<<dim3(NSRV / 16, 2), 128, 0, stream>>>(
        agg_s, hs_bf[cur],
        WlT + (size_t)(l * 2 + 1) * HDIM * HDIM,
        WrT + (size_t)(l * 2 + 1) * HDIM * HDIM,
        bl + (size_t)(l * 2 + 1) * HDIM,
        last ? out_hs : nullptr, last ? nullptr : hs_bf[nxt]);
    cur = nxt;
  }
  (void)in_sizes; (void)n_in; (void)out_size; (void)ws_size;
}
